// MultiHeadSelfAttention_17033840296514
// MI455X (gfx1250) — matrix operand order repaired and checked by reading, not compiled, not run on hardware
//
#include <hip/hip_runtime.h>
#include <math.h>

// ---------- CDNA5 (gfx1250) wave32 WMMA types ----------
typedef __attribute__((ext_vector_type(16))) _Float16 v16h;
typedef __attribute__((ext_vector_type(8)))  _Float16 v8h;
typedef __attribute__((ext_vector_type(8)))  float    v8f;
typedef __attribute__((ext_vector_type(4)))  float    v4f;

#define EMBED  1024
#define NHEADS 16
#define HEADD  64
#define SEQ    2048
#define BATCH  2

static __device__ __forceinline__ v16h hcat(v8h lo, v8h hi) {
  return __builtin_shufflevector(lo, hi, 0,1,2,3,4,5,6,7,8,9,10,11,12,13,14,15);
}

// A-operand fragment (16x32 f16): halves 0..7 = A[row][k0+8h..+7],
//                                 halves 8..15 = A[row][k0+16+8h..+7]
static __device__ __forceinline__ v16h ldA16(const _Float16* __restrict__ base,
                                             unsigned row, unsigned ld,
                                             unsigned k0, unsigned h) {
  const _Float16* p = base + row * ld + k0 + 8u * h;
  return hcat(*(const v8h*)p, *(const v8h*)(p + 16));
}

// B-operand fragment (32x16 f16, NT): halves 0..7 = B[col][k0+8h..+7], halves 8..15 = B[col][k0+16+8h..+7]
static __device__ __forceinline__ v16h ldB16(const _Float16* __restrict__ base,
                                             unsigned col, unsigned ld,
                                             unsigned k0, unsigned h) {
  const _Float16* p = base + col * ld + k0 + 8u * h;
  return hcat(*(const v8h*)p, *(const v8h*)(p + 16));
}

// ============================================================
// f32 -> f16 conversion (memory-bound prepass, 8 elems/thread)
// ============================================================
__global__ __launch_bounds__(256)
void k_cvt(const float* __restrict__ src, _Float16* __restrict__ dst, unsigned n) {
  unsigned i = (blockIdx.x * 256u + threadIdx.x) * 8u;
  if (i >= n) return;
  v4f x0 = *(const v4f*)(src + i);
  v4f x1 = *(const v4f*)(src + i + 4);
  v8h o;
#pragma unroll
  for (int j = 0; j < 4; ++j) { o[j] = (_Float16)x0[j]; o[4 + j] = (_Float16)x1[j]; }
  *(v8h*)(dst + i) = o;
}

// ============================================================
// Shared 32x64-per-wave NT GEMM core, ping-pong double buffered
// (no rotation copies). Accumulates into c[2][4].
// ============================================================
#define GEMM_LOAD(buf, kk)                                               \
  do {                                                                   \
    a0##buf = ldA16(Amat, ar0, EMBED, (kk), h);                          \
    a1##buf = ldA16(Amat, ar1, EMBED, (kk), h);                          \
    _Pragma("unroll")                                                    \
    for (int t = 0; t < 4; ++t)                                          \
      b##buf[t] = ldB16(Bmat, n0 + t * 16u + ln, EMBED, (kk), h);        \
  } while (0)

#define GEMM_MMA(buf)                                                    \
  do {                                                                   \
    _Pragma("unroll")                                                    \
    for (int t = 0; t < 4; ++t) {                                        \
      c[0][t] = __builtin_amdgcn_wmma_f32_16x16x32_f16(                  \
          false, a0##buf, false, b##buf[t], (short)0, c[0][t], false, false); \
      c[1][t] = __builtin_amdgcn_wmma_f32_16x16x32_f16(                  \
          false, a1##buf, false, b##buf[t], (short)0, c[1][t], false, false); \
    }                                                                    \
  } while (0)

#define GEMM_BODY()                                                      \
  v16h a0A, a1A, bA[4], a0B, a1B, bB[4];                                 \
  GEMM_LOAD(A, 0u);                                                      \
  unsigned k0 = 0;                                                       \
  _Pragma("unroll 1")                                                    \
  for (; k0 + 64u < EMBED; k0 += 64u) {                                  \
    GEMM_LOAD(B, k0 + 32u);                                              \
    GEMM_MMA(A);                                                         \
    GEMM_LOAD(A, k0 + 64u);                                              \
    GEMM_MMA(B);                                                         \
  }                                                                      \
  GEMM_LOAD(B, k0 + 32u);                                                \
  GEMM_MMA(A);                                                           \
  GEMM_MMA(B);

// ============================================================
// Kernel 1: QKV projection  C[4096,3072] = Xh @ Wh^T + b
// Epilogue: Q (pre-scaled by 1/8), K -> [bh][s][d];  V -> transposed [bh][d][s].
// ============================================================
__global__ __launch_bounds__(128)
void k_qkv(const _Float16* __restrict__ Xh, const _Float16* __restrict__ Wh,
           const float* __restrict__ bias,
           _Float16* __restrict__ qf, _Float16* __restrict__ kf,
           _Float16* __restrict__ vtf)
{
  const unsigned lane = threadIdx.x & 31u;
  const unsigned wave = threadIdx.x >> 5;
  const unsigned h    = lane >> 4;
  const unsigned ln   = lane & 15u;
  const unsigned m0   = blockIdx.x * 128u + wave * 32u;
  const unsigned n0   = blockIdx.y * 64u;
  const unsigned ar0  = m0 + ln, ar1 = m0 + 16u + ln;
  const _Float16* Amat = Xh;
  const _Float16* Bmat = Wh;

  v8f c[2][4] = {};
  GEMM_BODY();

  // Epilogue: bias + scatter (Q scaled by 1/sqrt(d)=0.125)
#pragma unroll
  for (int t = 0; t < 4; ++t) {
    const unsigned n   = n0 + t * 16u + ln;
    const float bval   = bias[n];
    const unsigned reg = n >> 10;          // 0=Q 1=K 2=V (uniform per tile)
    const unsigned nn  = n & 1023u;
    const unsigned head = nn >> 6, d = nn & 63u;
#pragma unroll
    for (int mi = 0; mi < 2; ++mi) {
#pragma unroll
      for (int r = 0; r < 8; ++r) {
        const unsigned m  = m0 + mi * 16u + r + 8u * h;
        const unsigned bb = m >> 11, s = m & 2047u;
        const unsigned bh = bb * NHEADS + head;
        const float v = c[mi][t][r] + bval;
        if (reg == 0)      qf [(bh * SEQ + s) * HEADD + d] = (_Float16)(v * 0.125f);
        else if (reg == 1) kf [(bh * SEQ + s) * HEADD + d] = (_Float16)v;
        else               vtf[(bh * HEADD + d) * SEQ + s] = (_Float16)v;
      }
    }
  }
}

// ============================================================
// Kernel 2: causal flash attention per (batch*head).
// Block = 64 query rows (4 waves x 16). KV blocks of 64.
// ============================================================
__global__ __launch_bounds__(128)
void k_attn(const _Float16* __restrict__ qf, const _Float16* __restrict__ kf,
            const _Float16* __restrict__ vtf, _Float16* __restrict__ of)
{
  __shared__ _Float16 Plds[4 * 16 * 64];   // per-wave P staging (C-layout -> A-layout)

  const unsigned lane = threadIdx.x & 31u;
  const unsigned wave = threadIdx.x >> 5;
  const unsigned h    = lane >> 4;
  const unsigned ln   = lane & 15u;
  const unsigned qblock = blockIdx.x;
  const unsigned bh     = blockIdx.y;
  const unsigned bb = bh >> 4, head = bh & 15u;
  const unsigned q0 = qblock * 64u + wave * 16u;

  const _Float16* qbase = qf  + bh * (SEQ * HEADD);
  const _Float16* kbase = kf  + bh * (SEQ * HEADD);
  const _Float16* vbase = vtf + bh * (HEADD * SEQ);
  _Float16* Pw = Plds + wave * 16 * 64;

  // Q fragments (A-operand, pre-scaled by 1/8): 2 K-chunks over d=64
  v16h aq[2];
#pragma unroll
  for (int cc = 0; cc < 2; ++cc) aq[cc] = ldA16(qbase, q0 + ln, HEADD, cc * 32u, h);

  v8f o[4] = {};
  float mrow[8], lrow[8];
#pragma unroll
  for (int r = 0; r < 8; ++r) { mrow[r] = -1e30f; lrow[r] = 0.f; }

  const unsigned nkb = qblock + 1u;   // wave-uniform causal trip count
  for (unsigned kb = 0; kb < nkb; ++kb) {
    const unsigned kv0 = kb * 64u;

    // ---- S = (Q/8) K^T : 4 N-tiles x 2 K-chunks ----
    float sv[4][8];
#pragma unroll
    for (int t = 0; t < 4; ++t) {
      const unsigned kvcol = kv0 + t * 16u + ln;
      v8f sacc = {};
#pragma unroll
      for (int cc = 0; cc < 2; ++cc) {
        v16h bk = ldB16(kbase, kvcol, HEADD, cc * 32u, h);
        sacc = __builtin_amdgcn_wmma_f32_16x16x32_f16(false, aq[cc], false, bk,
                                                      (short)0, sacc, false, false);
      }
#pragma unroll
      for (int r = 0; r < 8; ++r) sv[t][r] = sacc[r];
    }

    // ---- causal mask (diagonal block only) ----
    if (kb == nkb - 1u) {
#pragma unroll
      for (int t = 0; t < 4; ++t) {
        const unsigned kvcol = kv0 + t * 16u + ln;
#pragma unroll
        for (int r = 0; r < 8; ++r)
          if (kvcol > q0 + r + 8u * h) sv[t][r] = -1e30f;
      }
    }

    // ---- online softmax (row lives across 16 lanes of one half-wave) ----
    float alpha[8], pv[4][8];
#pragma unroll
    for (int r = 0; r < 8; ++r) {
      float mx = sv[0][r];
#pragma unroll
      for (int t = 1; t < 4; ++t) mx = fmaxf(mx, sv[t][r]);
#pragma unroll
      for (int mk = 1; mk < 16; mk <<= 1) mx = fmaxf(mx, __shfl_xor(mx, mk, 32));
      const float mnew = fmaxf(mrow[r], mx);
      alpha[r] = __expf(mrow[r] - mnew);
      float rs = 0.f;
#pragma unroll
      for (int t = 0; t < 4; ++t) {
        const float p = __expf(sv[t][r] - mnew);
        pv[t][r] = p; rs += p;
      }
#pragma unroll
      for (int mk = 1; mk < 16; mk <<= 1) rs += __shfl_xor(rs, mk, 32);
      lrow[r] = lrow[r] * alpha[r] + rs;
      mrow[r] = mnew;
    }
#pragma unroll
    for (int t = 0; t < 4; ++t)
#pragma unroll
      for (int r = 0; r < 8; ++r) o[t][r] *= alpha[r];

    // ---- stage P (C layout) to LDS, re-read in A layout (same-wave DS in-order) ----
#pragma unroll
    for (int t = 0; t < 4; ++t)
#pragma unroll
      for (int r = 0; r < 8; ++r)
        Pw[(r + 8u * h) * 64u + t * 16u + ln] = (_Float16)pv[t][r];
    __asm__ volatile("" ::: "memory");

    v16h pa[2];
#pragma unroll
    for (int cc = 0; cc < 2; ++cc) {
      const _Float16* pp = Pw + ln * 64u + cc * 32u + 8u * h;
      pa[cc] = hcat(*(const v8h*)pp, *(const v8h*)(pp + 16));
    }

    // ---- O += P V : 4 d-tiles x 2 KV-chunks (V^T gives contiguous B frags) ----
#pragma unroll
    for (int t = 0; t < 4; ++t) {
      const unsigned dcol = t * 16u + ln;
#pragma unroll
      for (int cc = 0; cc < 2; ++cc) {
        v16h bv = ldB16(vbase, dcol, SEQ, kv0 + cc * 32u, h);
        o[t] = __builtin_amdgcn_wmma_f32_16x16x32_f16(false, pa[cc], false, bv,
                                                      (short)0, o[t], false, false);
      }
    }
    __asm__ volatile("" ::: "memory");
  }

  // ---- normalize + store O as f16 [row][embed] for the output GEMM ----
#pragma unroll
  for (int r = 0; r < 8; ++r) {
    const float inv = 1.f / lrow[r];
    const unsigned q = q0 + r + 8u * h;
    const unsigned rowoff = (bb * SEQ + q) * EMBED + head * HEADD;
#pragma unroll
    for (int t = 0; t < 4; ++t)
      of[rowoff + t * 16u + ln] = (_Float16)(o[t][r] * inv);
  }
}

// ============================================================
// Kernel 3: output projection  Y[4096,1024] = O_f16 @ Wout_f16^T + b (f32 out)
// ============================================================
__global__ __launch_bounds__(128)
void k_oproj(const _Float16* __restrict__ Aof, const _Float16* __restrict__ Wh,
             const float* __restrict__ bias, float* __restrict__ Y)
{
  const unsigned lane = threadIdx.x & 31u;
  const unsigned wave = threadIdx.x >> 5;
  const unsigned h    = lane >> 4;
  const unsigned ln   = lane & 15u;
  const unsigned m0   = blockIdx.x * 128u + wave * 32u;
  const unsigned n0   = blockIdx.y * 64u;
  const unsigned ar0  = m0 + ln, ar1 = m0 + 16u + ln;
  const _Float16* Amat = Aof;
  const _Float16* Bmat = Wh;

  v8f c[2][4] = {};
  GEMM_BODY();

#pragma unroll
  for (int t = 0; t < 4; ++t) {
    const unsigned n = n0 + t * 16u + ln;
    const float bval = bias[n];
#pragma unroll
    for (int mi = 0; mi < 2; ++mi) {
#pragma unroll
      for (int r = 0; r < 8; ++r) {
        const unsigned m = m0 + mi * 16u + r + 8u * h;
        Y[m * EMBED + n] = c[mi][t][r] + bval;
      }
    }
  }
}

// ============================================================
extern "C" void kernel_launch(void* const* d_in, const int* in_sizes, int n_in,
                              void* d_out, int out_size, void* d_ws, size_t ws_size,
                              hipStream_t stream)
{
  const float* x     = (const float*)d_in[0];
  const float* W_qkv = (const float*)d_in[1];
  const float* b_qkv = (const float*)d_in[2];
  const float* W_out = (const float*)d_in[3];
  const float* b_out = (const float*)d_in[4];
  float* out = (float*)d_out;

  // Workspace (f16, all 16B aligned):
  //  qf|kf|vtf|of : 4 x 8MB,  xh 8MB, wqkvh 6MB, wouth 2MB  => 48MB total
  char* ws = (char*)d_ws;
  const size_t HB = (size_t)BATCH * NHEADS * SEQ * HEADD * sizeof(_Float16); // 8MB
  _Float16* qf    = (_Float16*)(ws);
  _Float16* kf    = (_Float16*)(ws + HB);
  _Float16* vtf   = (_Float16*)(ws + 2 * HB);
  _Float16* of    = (_Float16*)(ws + 3 * HB);
  _Float16* xh    = (_Float16*)(ws + 4 * HB);
  _Float16* wqkvh = (_Float16*)(ws + 5 * HB);
  _Float16* wouth = (_Float16*)(ws + 5 * HB + (size_t)3 * EMBED * EMBED * sizeof(_Float16));

  const unsigned nx = BATCH * SEQ * EMBED;        // 4M
  const unsigned nwq = 3 * EMBED * EMBED;         // 3M
  const unsigned nwo = EMBED * EMBED;             // 1M
  k_cvt<<<nx  / 2048, 256, 0, stream>>>(x,     xh,    nx);
  k_cvt<<<nwq / 2048, 256, 0, stream>>>(W_qkv, wqkvh, nwq);
  k_cvt<<<nwo / 2048, 256, 0, stream>>>(W_out, wouth, nwo);

  k_qkv  <<<dim3(32, 48), 128, 0, stream>>>(xh, wqkvh, b_qkv, qf, kf, vtf);
  k_attn <<<dim3(SEQ / 64, BATCH * NHEADS), 128, 0, stream>>>(qf, kf, vtf, of);
  k_oproj<<<dim3(32, 16), 128, 0, stream>>>(of, wouth, b_out, out);
}
